// StandardKernelModel_89412629168320
// MI455X (gfx1250) — matrix operand order repaired and checked by reading, not compiled, not run on hardware
//
#include <hip/hip_runtime.h>
#include <stdint.h>

typedef __attribute__((ext_vector_type(16))) __bf16 v16bf;
typedef __attribute__((ext_vector_type(8)))  __bf16 v8bf;
typedef __attribute__((ext_vector_type(4)))  __bf16 v4bf;
typedef __attribute__((ext_vector_type(8)))  float  v8f;

#define N_PTS 8192
#define M_CTR 8192
#define DIM   512
#define YDIM  256

#define ROWS_PER_BLK 64     // 4 waves x 16 rows
#define MCHUNK 32           // centers per inner iteration (one WMMA K-step for GEMM2)
#define MSPLIT 4            // grid.y split of the M reduction
#define MRANGE (M_CTR / MSPLIT)

// LDS row strides (in bf16 elements), padded to spread banks
#define XS_LD 520           // 512 + 8
#define ZS_LD 520
#define WT_LD 40            // 32 + 8
#define KS_LD 40

// ---------------------------------------------------------------------------
// Pre-pass: f32 rows -> bf16 rows + squared-norm per row (one wave per row)
// ---------------------------------------------------------------------------
__global__ void rows_to_bf16_sq(const float* __restrict__ src,
                                __bf16* __restrict__ dst,
                                float* __restrict__ sq) {
  const int row  = blockIdx.x * 8 + (threadIdx.x >> 5);
  const int lane = threadIdx.x & 31;
  const float* p = src + (size_t)row * DIM;
  __bf16*      q = dst + (size_t)row * DIM;
  float s = 0.0f;
#pragma unroll
  for (int it = 0; it < 4; ++it) {
    const int c = it * 128 + lane * 4;
    float4 v = *(const float4*)(p + c);
    s += v.x * v.x + v.y * v.y + v.z * v.z + v.w * v.w;
    v4bf b;
    b[0] = (__bf16)v.x; b[1] = (__bf16)v.y; b[2] = (__bf16)v.z; b[3] = (__bf16)v.w;
    *(v4bf*)(q + c) = b;
  }
#pragma unroll
  for (int off = 16; off > 0; off >>= 1) s += __shfl_xor(s, off, 32);
  if (lane == 0) sq[row] = s;
}

// W [M x Y] f32 -> Wt [Y x M] bf16 (transposed so GEMM2 B-frags are contiguous)
__global__ void w_transpose_bf16(const float* __restrict__ w,
                                 __bf16* __restrict__ wt) {
  const int idx = blockIdx.x * 256 + threadIdx.x;   // coalesced read of W
  const int k = idx >> 8;                           // row of W (center index)
  const int n = idx & 255;                          // col of W (output dim)
  wt[(size_t)n * M_CTR + k] = (__bf16)w[idx];
}

__global__ void zero_f32(float* __restrict__ p) {
  p[(size_t)blockIdx.x * 256 + threadIdx.x] = 0.0f;
}

// ---------------------------------------------------------------------------
// Fused kernel: S = Xb.Zb^T (bf16 WMMA, f32 acc) -> K = exp(-sqrt(d2)/10)
//               -> P += K.W (bf16 WMMA, f32 acc), K never leaves LDS.
// ---------------------------------------------------------------------------
__launch_bounds__(128, 1)
__global__ void laplace_fused(const __bf16* __restrict__ Xb,
                              const __bf16* __restrict__ Zb,
                              const __bf16* __restrict__ Wt,
                              const float*  __restrict__ xsq,
                              const float*  __restrict__ zsq,
                              float* __restrict__ out) {
  __shared__ __align__(16) __bf16 Xs [ROWS_PER_BLK * XS_LD];  // 66.5 KB
  __shared__ __align__(16) __bf16 Zs [MCHUNK * ZS_LD];        // 33.3 KB
  __shared__ __align__(16) __bf16 Wts[YDIM * WT_LD];          // 20.5 KB
  __shared__ __align__(16) __bf16 Kst[4 * 16 * KS_LD];        //  5.0 KB (wave-private slabs)
  __shared__ float xsqs[ROWS_PER_BLK];
  __shared__ float zsqs[MCHUNK];

  const int tid  = threadIdx.x;
  const int lane = tid & 31;
  const int wid  = tid >> 5;
  const int row0 = blockIdx.x * ROWS_PER_BLK;
  const int mc0  = blockIdx.y * MRANGE;

  const int wm  = wid * 16;        // wave's first row within block
  const int hi  = lane >> 4;       // lane half (WMMA layout select)
  const int l15 = lane & 15;

  // ---- one-time async stage of this block's X rows (bf16, 1 KB/row = 2 segs)
#pragma unroll
  for (int i = 0; i < 32; ++i) {
    const int s = wid * 32 + i;            // 128 segments of 512B
    const int r = s >> 1, h = s & 1;
    uint64_t ga = (uint64_t)(uintptr_t)(Xb + (size_t)(row0 + r) * DIM + h * 256 + lane * 8);
    uint32_t la = (uint32_t)(uintptr_t)(&Xs[r * XS_LD + h * 256 + lane * 8]);
    asm volatile("global_load_async_to_lds_b128 %0, %1, off" :: "v"(la), "v"(ga) : "memory");
  }
  if (tid < ROWS_PER_BLK) xsqs[tid] = xsq[row0 + tid];
  asm volatile("s_wait_asynccnt 0" ::: "memory");
  __syncthreads();

  v8f acc[16] = {};                // 16 rows x 256 cols of P, f32

  for (int mc = mc0; mc < mc0 + MRANGE; mc += MCHUNK) {
    // ---- async stage Z chunk: 32 rows x 1KB = 64 segments (16 per wave)
#pragma unroll
    for (int i = 0; i < 16; ++i) {
      const int s = wid * 16 + i;
      const int r = s >> 1, h = s & 1;
      uint64_t ga = (uint64_t)(uintptr_t)(Zb + (size_t)(mc + r) * DIM + h * 256 + lane * 8);
      uint32_t la = (uint32_t)(uintptr_t)(&Zs[r * ZS_LD + h * 256 + lane * 8]);
      asm volatile("global_load_async_to_lds_b128 %0, %1, off" :: "v"(la), "v"(ga) : "memory");
    }
    // ---- async stage Wt chunk: 256 rows x 64B (4 lanes per row, 8 rows/instr)
#pragma unroll
    for (int i = 0; i < 8; ++i) {
      const int s = wid * 8 + i;                 // 0..31
      const int n = s * 8 + (lane >> 2);         // output-dim row 0..255
      const int b = (lane & 3) * 8;              // element offset inside 64B row
      uint64_t ga = (uint64_t)(uintptr_t)(Wt + (size_t)n * M_CTR + mc + b);
      uint32_t la = (uint32_t)(uintptr_t)(&Wts[n * WT_LD + b]);
      asm volatile("global_load_async_to_lds_b128 %0, %1, off" :: "v"(la), "v"(ga) : "memory");
    }
    if (tid < MCHUNK) zsqs[tid] = zsq[mc + tid];
    asm volatile("s_wait_asynccnt 0" ::: "memory");
    __syncthreads();

    // ---- GEMM1: S[16x32] = X[16x512] . Z_chunk^T, bf16 WMMA, K = 512
    v8f s0 = {}, s1 = {};
    {
      const int akb = hi ? 8 : 0;     // A frag: K split {0-7,16-23} / {8-15,24-31}
      const int bkb = hi ? 8 : 0;     // B frag: K split {0-7,16-23} / {8-15,24-31}
#pragma unroll
      for (int ks = 0; ks < DIM; ks += 32) {
        union { v16bf v; v8bf h[2]; } ua, ub0, ub1;
        ua.h[0]  = *(const v8bf*)&Xs[(wm + l15) * XS_LD + ks + akb];
        ua.h[1]  = *(const v8bf*)&Xs[(wm + l15) * XS_LD + ks + akb + 16];
        ub0.h[0] = *(const v8bf*)&Zs[l15 * ZS_LD + ks + bkb];
        ub0.h[1] = *(const v8bf*)&Zs[l15 * ZS_LD + ks + bkb + 16];
        ub1.h[0] = *(const v8bf*)&Zs[(16 + l15) * ZS_LD + ks + bkb];
        ub1.h[1] = *(const v8bf*)&Zs[(16 + l15) * ZS_LD + ks + bkb + 16];
        s0 = __builtin_amdgcn_wmma_f32_16x16x32_bf16(false, ua.v, false, ub0.v,
                                                     (short)0, s0, false, false);
        s1 = __builtin_amdgcn_wmma_f32_16x16x32_bf16(false, ua.v, false, ub1.v,
                                                     (short)0, s1, false, false);
      }
    }

    // ---- elementwise Laplacian kernel + bf16 K-slab to wave-private LDS
    __bf16* kbase = &Kst[wid * 16 * KS_LD];
#pragma unroll
    for (int i = 0; i < 8; ++i) {
      const int ml = i + hi * 8;                 // local row (C-matrix layout)
      const float xq = xsqs[wm + ml];
      {
        float d2 = xq + zsqs[l15] - 2.0f * s0[i];
        float dist = __builtin_sqrtf(fmaxf(d2, 0.0f));
        kbase[ml * KS_LD + l15] = (__bf16)__expf(dist * -0.1f);
      }
      {
        float d2 = xq + zsqs[16 + l15] - 2.0f * s1[i];
        float dist = __builtin_sqrtf(fmaxf(d2, 0.0f));
        kbase[ml * KS_LD + 16 + l15] = (__bf16)__expf(dist * -0.1f);
      }
    }

    // ---- GEMM2: P[16x256] += K[16x32] . W_chunk[32x256], single K-step
    {
      const int akb = hi ? 8 : 0;
      const int bkb = hi ? 8 : 0;
      union { v16bf v; v8bf h[2]; } ua;
      ua.h[0] = *(const v8bf*)&kbase[l15 * KS_LD + akb];
      ua.h[1] = *(const v8bf*)&kbase[l15 * KS_LD + akb + 16];
#pragma unroll
      for (int t = 0; t < 16; ++t) {
        union { v16bf v; v8bf h[2]; } ub;
        ub.h[0] = *(const v8bf*)&Wts[(t * 16 + l15) * WT_LD + bkb];
        ub.h[1] = *(const v8bf*)&Wts[(t * 16 + l15) * WT_LD + bkb + 16];
        acc[t] = __builtin_amdgcn_wmma_f32_16x16x32_bf16(false, ua.v, false, ub.v,
                                                         (short)0, acc[t], false, false);
      }
    }
    __syncthreads();   // protect Zs/Wts before next iteration's async stores
  }

  // ---- accumulate partial P into global output (M was split MSPLIT ways)
#pragma unroll
  for (int t = 0; t < 16; ++t) {
#pragma unroll
    for (int i = 0; i < 8; ++i) {
      atomicAdd(&out[(size_t)(row0 + wm + i + hi * 8) * YDIM + t * 16 + l15], acc[t][i]);
    }
  }
}

// ---------------------------------------------------------------------------
extern "C" void kernel_launch(void* const* d_in, const int* in_sizes, int n_in,
                              void* d_out, int out_size, void* d_ws, size_t ws_size,
                              hipStream_t stream) {
  const float* x = (const float*)d_in[0];   // [8192, 512]
  const float* z = (const float*)d_in[1];   // [8192, 512]
  const float* w = (const float*)d_in[2];   // [8192, 256]
  float* out = (float*)d_out;               // [8192, 256]

  char* ws = (char*)d_ws;                   // needs ~21.0 MB
  float*  xsq = (float*)ws;  ws += (size_t)N_PTS * 4;
  float*  zsq = (float*)ws;  ws += (size_t)M_CTR * 4;
  __bf16* Xb  = (__bf16*)ws; ws += (size_t)N_PTS * DIM * 2;
  __bf16* Zb  = (__bf16*)ws; ws += (size_t)M_CTR * DIM * 2;
  __bf16* Wt  = (__bf16*)ws;                // [YDIM][M_CTR]

  rows_to_bf16_sq<<<N_PTS / 8, 256, 0, stream>>>(x, Xb, xsq);
  rows_to_bf16_sq<<<M_CTR / 8, 256, 0, stream>>>(z, Zb, zsq);
  w_transpose_bf16<<<(M_CTR * YDIM) / 256, 256, 0, stream>>>(w, Wt);
  zero_f32<<<(N_PTS * YDIM) / 256, 256, 0, stream>>>(out);

  dim3 grid(N_PTS / ROWS_PER_BLK, MSPLIT);
  laplace_fused<<<grid, 128, 0, stream>>>(Xb, Zb, Wt, xsq, zsq, out);
}
